// EdgeFormerSelfAttention_1743756722832
// MI455X (gfx1250) — matrix operand order repaired and checked by reading, not compiled, not run on hardware
//
#include <hip/hip_runtime.h>

typedef __attribute__((ext_vector_type(16))) _Float16 v16h;
typedef __attribute__((ext_vector_type(8)))  _Float16 v8h;
typedef __attribute__((ext_vector_type(8)))  float    v8f;

#define S_LEN 4096
#define HID   1024
#define NH    16
#define DH    64
#define HALF_WIN 256

// ---------------------------------------------------------------------------
// Fragment loaders following the CDNA5 WMMA VGPR layout tables (CDNA5 ISA):
//  A 16x32 f16: lanes 0-15 (row M=lane) hold K={0..7,16..23}; lanes 16-31 hold
//               K={8..15,24..31}  -> two contiguous 16B loads per lane.
//  B 32x16 f16: lanes 0-15 (col N=lane) hold K={0..7,16..23}; lanes 16-31 hold
//               K={8..15,24..31} -> two contiguous 16B loads per lane.
// `base` points at the tile origin (row0/col0, k0 already applied by caller),
// `ld` is the leading dimension in halfs along the lane-major axis.
// ---------------------------------------------------------------------------
__device__ __forceinline__ v16h load_fragA(const _Float16* base, int ld, int lane) {
  int m = lane & 15, kh = lane >> 4;
  const _Float16* p = base + (size_t)m * ld + kh * 8;
  union { v16h v; v8h h[2]; } u;
  u.h[0] = *(const v8h*)(p);
  u.h[1] = *(const v8h*)(p + 16);
  return u.v;
}

__device__ __forceinline__ v16h load_fragB(const _Float16* base, int ld, int lane) {
  int n = lane & 15, kh = lane >> 4;
  const _Float16* p = base + (size_t)n * ld + kh * 8;
  union { v16h v; v8h h[2]; } u;
  u.h[0] = *(const v8h*)(p);
  u.h[1] = *(const v8h*)(p + 16);
  return u.v;
}

__device__ __forceinline__ v8f wmma_f16(v16h a, v16h b, v8f c) {
  return __builtin_amdgcn_wmma_f32_16x16x32_f16(false, a, false, b, (short)0, c,
                                                false, false);
}

// ---------------------------------------------------------------------------
// fp32 -> f16 conversion (hidden states + weights)
// ---------------------------------------------------------------------------
__global__ void cvt_f32_to_f16(const float* __restrict__ src,
                               _Float16* __restrict__ dst, int n) {
  int i = blockIdx.x * blockDim.x + threadIdx.x;
  if (i < n) dst[i] = (_Float16)src[i];
}

// ---------------------------------------------------------------------------
// QKV projection: out = X @ W^T + b.
// Register-blocked: one wave per 64x64 output tile. Per K-step of 32:
//   4 A-frags + 4 B-frags (16 B128 loads) feed 16 WMMAs -> 1 load/WMMA,
//   ~32 FLOP/byte from L2 (vs ~8 with 16x16 tiles).
//   mat 0 -> Q stored [S,1024] f16, pre-scaled by 1/sqrt(64)
//   mat 1 -> K stored [S,1024] f16
//   mat 2 -> V stored transposed [1024,S] f16 (so PV B-frags are contiguous)
// ---------------------------------------------------------------------------
__global__ __launch_bounds__(256) void qkv_gemm(
    const _Float16* __restrict__ Xh,
    const _Float16* __restrict__ Wqh, const _Float16* __restrict__ Wkh,
    const _Float16* __restrict__ Wvh,
    const float* __restrict__ bq, const float* __restrict__ bk,
    const float* __restrict__ bv,
    _Float16* __restrict__ Qh, _Float16* __restrict__ Kh,
    _Float16* __restrict__ Vt) {
  int lane = threadIdx.x & 31, wid = threadIdx.x >> 5;
  int w = blockIdx.x * 8 + wid;          // 3072 waves = 3 * 16 * 64 tiles
  int stile = w & 63;                    // 64 row-blocks of 64
  int otile = (w >> 6) & 15;             // 16 col-blocks of 64
  int mat = w >> 10;                     // 0=Q 1=K 2=V
  int s0 = stile * 64, o0 = otile * 64;

  const _Float16* Wm  = (mat == 0) ? Wqh : (mat == 1) ? Wkh : Wvh;
  const float*    bia = (mat == 0) ? bq  : (mat == 1) ? bk  : bv;

  const _Float16* arow = Xh + (size_t)s0 * HID;
  const _Float16* brow = Wm + (size_t)o0 * HID;

  v8f c[4][4];
#pragma unroll
  for (int i = 0; i < 4; ++i)
#pragma unroll
    for (int j = 0; j < 4; ++j) c[i][j] = (v8f){};

  for (int k = 0; k < HID; k += 32) {
    v16h a[4], b[4];
#pragma unroll
    for (int i = 0; i < 4; ++i)
      a[i] = load_fragA(arow + (size_t)(16 * i) * HID + k, HID, lane);
#pragma unroll
    for (int j = 0; j < 4; ++j)
      b[j] = load_fragB(brow + (size_t)(16 * j) * HID + k, HID, lane);
#pragma unroll
    for (int i = 0; i < 4; ++i)
#pragma unroll
      for (int j = 0; j < 4; ++j)
        c[i][j] = wmma_f16(a[i], b[j], c[i][j]);
  }

  int n = lane & 15, hh = lane >> 4;
#pragma unroll
  for (int j = 0; j < 4; ++j) {
    int o = o0 + 16 * j + n;
    float bb = bia[o];
#pragma unroll
    for (int i = 0; i < 4; ++i) {
#pragma unroll
      for (int r = 0; r < 8; ++r) {
        int s = s0 + 16 * i + r + 8 * hh;
        float val = c[i][j][r] + bb;
        if (mat == 0)      Qh[(size_t)s * HID + o] = (_Float16)(val * 0.125f);
        else if (mat == 1) Kh[(size_t)s * HID + o] = (_Float16)val;
        else               Vt[(size_t)o * S_LEN + s] = (_Float16)val;
      }
    }
  }
}

// ---------------------------------------------------------------------------
// Sliding-window attention, flash-style online softmax.
// One wave per (head, query-tile of 16 rows). Iterates 32-key tile pairs
// inside the +-256 window: 4 score WMMAs + softmax + 4 PV WMMAs per pair.
// Prefetches next tile's K/V rows (gfx1250 global_prefetch path).
// ---------------------------------------------------------------------------
__global__ __launch_bounds__(256) void swa_attn(
    const _Float16* __restrict__ Qh, const _Float16* __restrict__ Kh,
    const _Float16* __restrict__ Vt, const float* __restrict__ amask,
    float* __restrict__ out) {
  __shared__ __align__(16) _Float16 plds[8][16 * 32];  // per-wave P staging
  int lane = threadIdx.x & 31, wid = threadIdx.x >> 5;
  int w = blockIdx.x * 8 + wid;          // 4096 waves = 16 heads * 256 qtiles
  int head = w >> 8;
  int qt = w & 255;
  int q0 = qt * 16;
  int n = lane & 15, hh = lane >> 4;

  const _Float16* Qb = Qh + (size_t)q0 * HID + head * DH;
  v16h aq0 = load_fragA(Qb, HID, lane);        // d = 0..31
  v16h aq1 = load_fragA(Qb + 32, HID, lane);   // d = 32..63

  float mrow[8], srow[8];
  v8f acc0 = {}, acc1 = {}, acc2 = {}, acc3 = {};
#pragma unroll
  for (int r = 0; r < 8; ++r) { mrow[r] = -1e30f; srow[r] = 0.f; }

  int ktlo = qt - 16; if (ktlo < 0) ktlo = 0;
  int kthi = qt + 16; if (kthi > 255) kthi = 255;
  _Float16* pl = plds[wid];
  const _Float16* Vbase = Vt + (size_t)(head * DH) * S_LEN;

  for (int ktp = ktlo >> 1; ktp <= kthi >> 1; ++ktp) {
    int j0 = ktp * 32;

    // prefetch next 32-key tile of K and V^T toward the WGP
    if (j0 + 32 < S_LEN) {
      __builtin_prefetch(Kh + (size_t)(j0 + 32 + lane) * HID + head * DH, 0, 3);
      __builtin_prefetch(Vbase + (size_t)(lane * 2) * S_LEN + j0 + 32, 0, 3);
    }

    // ---- scores: two 16x16 tiles covering keys j0..j0+31 ----
    const _Float16* Kb = Kh + (size_t)j0 * HID + head * DH;
    v8f c0 = {}, c1 = {};
    c0 = wmma_f16(aq0, load_fragB(Kb, HID, lane), c0);
    c0 = wmma_f16(aq1, load_fragB(Kb + 32, HID, lane), c0);
    c1 = wmma_f16(aq0, load_fragB(Kb + 16 * HID, HID, lane), c1);
    c1 = wmma_f16(aq1, load_fragB(Kb + 16 * HID + 32, HID, lane), c1);

    int j_a = j0 + n, j_b = j0 + 16 + n;
    float am_a = amask[j_a], am_b = amask[j_b];
#pragma unroll
    for (int r = 0; r < 8; ++r) {
      int i = q0 + r + 8 * hh;
      int da = j_a - i; if (da < 0) da = -da;
      int db = j_b - i; if (db < 0) db = -db;
      float sa = c0[r] + am_a + (da > HALF_WIN ? -10000.f : 0.f);
      float sb = c1[r] + am_b + (db > HALF_WIN ? -10000.f : 0.f);
      // row max across the 16 lanes holding this row
      float tmax = fmaxf(sa, sb);
      for (int msk = 1; msk < 16; msk <<= 1)
        tmax = fmaxf(tmax, __shfl_xor(tmax, msk, 32));
      float mnew  = fmaxf(mrow[r], tmax);
      float scale = __expf(mrow[r] - mnew);
      float pa = __expf(sa - mnew);
      float pb = __expf(sb - mnew);
      float psum = pa + pb;
      for (int msk = 1; msk < 16; msk <<= 1)
        psum += __shfl_xor(psum, msk, 32);
      srow[r] = srow[r] * scale + psum;
      mrow[r] = mnew;
      acc0[r] *= scale; acc1[r] *= scale; acc2[r] *= scale; acc3[r] *= scale;
      // stage P (C-layout -> A-layout via LDS)
      int row = r + 8 * hh;
      pl[row * 32 + n]      = (_Float16)pa;
      pl[row * 32 + 16 + n] = (_Float16)pb;
    }
    v16h pA = load_fragA(pl, 32, lane);   // 16 q-rows x 32 keys

    // ---- PV: V^T [d][S] so K-dim (keys) is contiguous per lane ----
    const _Float16* Vb = Vbase + j0;
    acc0 = wmma_f16(pA, load_fragB(Vb + (size_t)0  * S_LEN, S_LEN, lane), acc0);
    acc1 = wmma_f16(pA, load_fragB(Vb + (size_t)16 * S_LEN, S_LEN, lane), acc1);
    acc2 = wmma_f16(pA, load_fragB(Vb + (size_t)32 * S_LEN, S_LEN, lane), acc2);
    acc3 = wmma_f16(pA, load_fragB(Vb + (size_t)48 * S_LEN, S_LEN, lane), acc3);
  }

#pragma unroll
  for (int r = 0; r < 8; ++r) {
    int i = q0 + r + 8 * hh;
    float inv = 1.f / srow[r];
    float* orow = out + (size_t)i * HID + head * DH + n;
    orow[0]  = acc0[r] * inv;
    orow[16] = acc1[r] * inv;
    orow[32] = acc2[r] * inv;
    orow[48] = acc3[r] * inv;
  }
}

// ---------------------------------------------------------------------------
// Launch
// ---------------------------------------------------------------------------
extern "C" void kernel_launch(void* const* d_in, const int* in_sizes, int n_in,
                              void* d_out, int out_size, void* d_ws,
                              size_t ws_size, hipStream_t stream) {
  (void)in_sizes; (void)n_in; (void)out_size; (void)ws_size;
  const float* X  = (const float*)d_in[0];
  const float* am = (const float*)d_in[1];
  const float* Wq = (const float*)d_in[2];
  const float* bq = (const float*)d_in[3];
  const float* Wk = (const float*)d_in[4];
  const float* bk = (const float*)d_in[5];
  const float* Wv = (const float*)d_in[6];
  const float* bv = (const float*)d_in[7];
  float* out = (float*)d_out;

  // workspace layout (halfs): X(4M) Wq(1M) Wk(1M) Wv(1M) Q(4M) K(4M) V^T(4M)
  // total ~38 MB
  _Float16* ws  = (_Float16*)d_ws;
  const size_t NX = (size_t)S_LEN * HID;   // 4,194,304
  const size_t NW = (size_t)HID * HID;     // 1,048,576
  _Float16* Xh  = ws;
  _Float16* Wqh = Xh + NX;
  _Float16* Wkh = Wqh + NW;
  _Float16* Wvh = Wkh + NW;
  _Float16* Qh  = Wvh + NW;
  _Float16* Kh  = Qh + NX;
  _Float16* Vt  = Kh + NX;

  cvt_f32_to_f16<<<(int)(NX / 256), 256, 0, stream>>>(X, Xh, (int)NX);
  cvt_f32_to_f16<<<(int)(NW / 256), 256, 0, stream>>>(Wq, Wqh, (int)NW);
  cvt_f32_to_f16<<<(int)(NW / 256), 256, 0, stream>>>(Wk, Wkh, (int)NW);
  cvt_f32_to_f16<<<(int)(NW / 256), 256, 0, stream>>>(Wv, Wvh, (int)NW);

  // 3 matrices * 64 s-blocks * 16 o-blocks = 3072 waves, 8 waves/block
  qkv_gemm<<<384, 256, 0, stream>>>(Xh, Wqh, Wkh, Wvh, bq, bk, bv, Qh, Kh, Vt);

  // 16 heads * 256 q-tiles = 4096 waves, 8 waves/block
  swa_attn<<<512, 256, 0, stream>>>(Qh, Kh, Vt, am, out);
}
